// MPScaledDotProductAttention_88613765251732
// MI455X (gfx1250) — matrix operand order repaired and checked by reading, not compiled, not run on hardware
//
#include <hip/hip_runtime.h>
#include <hip/hip_bf16.h>

typedef float     v8f  __attribute__((ext_vector_type(8)));
typedef float     v4f  __attribute__((ext_vector_type(4)));
typedef float     v2f  __attribute__((ext_vector_type(2)));
typedef _Float16  v16h __attribute__((ext_vector_type(16)));
typedef _Float16  v8h  __attribute__((ext_vector_type(8)));
typedef _Float16  v2h  __attribute__((ext_vector_type(2)));

#define SEQ   1024
#define HD    64
#define NBH   32      // B*H
#define NMASK 4
#define ROWS  16      // rows per workgroup
#define PSTR  1032    // probs LDS stride (halves) -> conflict-free b128 reads

// --------------------------------------------------------------------------
// Pre-pass: V [bh][s][d] (f32) -> V^T [bh][d][s] (f16), LDS-tiled 64x64
// transpose so both global sides are coalesced. Grid: NBH*16 blocks.
// --------------------------------------------------------------------------
__global__ __launch_bounds__(256) void vth_convert_kernel(
    const float* __restrict__ v, _Float16* __restrict__ vth) {
  __shared__ float ld[64][65];                // pad -> conflict-free transpose
  const int bh = blockIdx.x >> 4;
  const int s0 = (blockIdx.x & 15) << 6;
  const float* vb = v + ((size_t)bh << 16);   // HD*SEQ per (b,h)
  _Float16*    tb = vth + ((size_t)bh << 16);
  const int tid = threadIdx.x;
#pragma unroll
  for (int i = 0; i < 16; ++i) {              // coalesced f32 reads
    int flat = i * 256 + tid;
    int s = flat >> 6, d = flat & 63;
    ld[s][d] = vb[(size_t)(s0 + s) * HD + d];
  }
  __syncthreads();
#pragma unroll
  for (int i = 0; i < 8; ++i) {               // coalesced packed-f16 writes
    int flat = i * 512 + tid * 2;
    int d = flat >> 6, sl = flat & 63;
    v2h p;
    p[0] = (_Float16)ld[sl][d];
    p[1] = (_Float16)ld[sl + 1][d];
    *(v2h*)(tb + (size_t)d * SEQ + s0 + sl) = p;
  }
}

// --------------------------------------------------------------------------
// Fused attention: QK^T (f32 WMMA) -> LDS, coalesced NT attn write,
// masked softmax (register-cached), P (f16) x V^T (f16) WMMA -> output.
// Grid: NBH * (SEQ/ROWS) = 2048 blocks, 256 threads (8 wave32).
// --------------------------------------------------------------------------
__global__ __launch_bounds__(256) void attn_kernel(
    const float* __restrict__ q, const float* __restrict__ k,
    const float* __restrict__ mask, const _Float16* __restrict__ vth,
    float* __restrict__ out, float* __restrict__ attn) {

  __shared__ float    sc[ROWS * SEQ];          // 64 KB exact f32 scores
  __shared__ _Float16 pb[2 * ROWS * PSTR];     // 66 KB probs, 2 masks at a time

  const int tid  = threadIdx.x;
  const int wave = tid >> 5;
  const int lane = tid & 31;
  const int lo   = lane & 15;     // M (A/C) or N (B/C) index
  const int hi   = lane >> 4;     // half-wave select

  const int bh   = blockIdx.x >> 6;
  const int row0 = (blockIdx.x & 63) << 4;

  // ------------------- Phase 1: scores = (Q/8) K^T -------------------
  const float* qrow = q + ((size_t)bh * SEQ + row0 + lo) * HD;

#if __has_builtin(__builtin_amdgcn_wmma_f32_16x16x4_f32)
  // A-matrix 16x4 f32: lane(lo,hi) holds row lo, K = {4t+2hi, 4t+2hi+1}
  v2f aq[16];
#pragma unroll
  for (int t = 0; t < 16; ++t) {
    v2f x = *(const v2f*)(qrow + 4 * t + 2 * hi);
    aq[t] = x * 0.125f;                       // fold 1/TEMPERATURE into Q
  }
#else
  v16h aqh[2];
#pragma unroll
  for (int c = 0; c < 2; ++c)
#pragma unroll
    for (int j = 0; j < 8; ++j) {
      aqh[c][j]     = (_Float16)(qrow[32 * c + 8 * hi + j] * 0.125f);
      aqh[c][j + 8] = (_Float16)(qrow[32 * c + 16 + 8 * hi + j] * 0.125f);
    }
#endif

#pragma unroll 1
  for (int ct = 0; ct < 8; ++ct) {            // 8 column tiles per wave
    const int n0 = (wave * 8 + ct) << 4;
    const float* krow = k + ((size_t)bh * SEQ + n0 + lo) * HD;
    v8f acc = {};
#if __has_builtin(__builtin_amdgcn_wmma_f32_16x16x4_f32)
#pragma unroll
    for (int t = 0; t < 16; ++t) {
      v2f bk = *(const v2f*)(krow + 4 * t + 2 * hi);
      acc = __builtin_amdgcn_wmma_f32_16x16x4_f32(
          false, aq[t], false, bk, (short)0, acc, false, false);
    }
#else
#pragma unroll
    for (int c = 0; c < 2; ++c) {
      v16h bk;
#pragma unroll
      for (int j = 0; j < 16; ++j)
        bk[j] = (_Float16)krow[32 * c + 16 * hi + j];
      acc = __builtin_amdgcn_wmma_f32_16x16x32_f16(
          false, aqh[c], false, bk, (short)0, acc, false, false);
    }
#endif
    // C layout: element (m = r + 8*hi, n = lo) in acc[r]
    float* srow = sc + n0 + lo;
#pragma unroll
    for (int r = 0; r < 8; ++r) srow[(r + 8 * hi) * SEQ] = acc[r];
  }
  __syncthreads();

  // ---- Coalesced non-temporal attn write straight from the LDS tile ----
  {
    float* ab = attn + (size_t)bh * SEQ * SEQ + (size_t)row0 * SEQ;
#pragma unroll
    for (int i = 0; i < 16; ++i) {
      int f = (i * 256 + tid) * 4;
      v4f x = *(const v4f*)(sc + f);
      __builtin_nontemporal_store(x, (v4f*)(ab + f));
    }
  }

  // ------------- Phase 2: masked softmax + P*V, two masks per pass -------------
  const int srow_id = tid >> 4;   // 16 rows x 16 threads
  const int jj      = tid & 15;

  for (int pr = 0; pr < 2; ++pr) {
    for (int kl = 0; kl < 2; ++kl) {
      const int km = pr * 2 + kl;
      const float* mrow =
          mask + (size_t)km * SEQ * SEQ + (size_t)(row0 + srow_id) * SEQ;
      const float* s_r = sc + srow_id * SEQ;

      float val[64];                // register-cached masked logits
      float mx = -3.0e38f;
#pragma unroll
      for (int c = 0; c < 64; ++c) {
        int col = jj + 16 * c;      // coalesced global / conflict-free LDS
        float s = s_r[col];
        val[c] = (mrow[col] != 0.0f) ? s : (s - 10000.0f);
        mx = fmaxf(mx, val[c]);
      }
#pragma unroll
      for (int o = 8; o > 0; o >>= 1) mx = fmaxf(mx, __shfl_xor(mx, o, 32));

      float sum = 0.0f;
#pragma unroll
      for (int c = 0; c < 64; ++c) {
        val[c] = __expf(val[c] - mx);
        sum += val[c];
      }
#pragma unroll
      for (int o = 8; o > 0; o >>= 1) sum += __shfl_xor(sum, o, 32);
      float inv = 1.0f / sum;

      _Float16* p_r = pb + (kl * ROWS + srow_id) * PSTR;
#pragma unroll
      for (int c = 0; c < 64; ++c)
        p_r[jj + 16 * c] = (_Float16)(val[c] * inv);
    }
    __syncthreads();

    // P*V: wave -> (mask kl = wave>>2, output columns d0 = (wave&3)*16)
    {
      const int kl = wave >> 2;
      const int d0 = (wave & 3) << 4;
      const _Float16* vcol = vth + ((size_t)bh * HD + d0 + lo) * SEQ;
      const _Float16* prow = pb + (kl * ROWS + lo) * PSTR;
      v8f acc = {};
#pragma unroll 4
      for (int kc = 0; kc < SEQ; kc += 32) {
        // A 16x32 f16: halves[0..7]=K(kc+8hi..), halves[8..15]=K(kc+16+8hi..)
        v8h a0 = *(const v8h*)(prow + kc + 8 * hi);
        v8h a1 = *(const v8h*)(prow + kc + 8 * hi + 16);
        v16h A = __builtin_shufflevector(a0, a1, 0, 1, 2, 3, 4, 5, 6, 7, 8, 9,
                                         10, 11, 12, 13, 14, 15);
        // B 32x16 f16: lane holds column d0+lo, halves[0..7]=K(kc+8hi..), halves[8..15]=K(kc+16+8hi..)
        v8h b0 = *(const v8h*)(vcol + kc + 8 * hi);
        v8h b1 = *(const v8h*)(vcol + kc + 8 * hi + 16);
        v16h B = __builtin_shufflevector(b0, b1, 0, 1, 2, 3, 4, 5, 6, 7, 8, 9,
                                         10, 11, 12, 13, 14, 15);
        acc = __builtin_amdgcn_wmma_f32_16x16x32_f16(
            false, A, false, B, (short)0, acc, false, false);
      }
      const int km = pr * 2 + kl;
      float* orow =
          out + ((size_t)(km * NBH + bh) * SEQ + row0) * HD + d0 + lo;
#pragma unroll
      for (int r = 0; r < 8; ++r)
        __builtin_nontemporal_store(acc[r], orow + (size_t)(r + 8 * hi) * HD);
    }
    __syncthreads();   // probs buffer reused by next mask pair
  }
}

extern "C" void kernel_launch(void* const* d_in, const int* in_sizes, int n_in,
                              void* d_out, int out_size, void* d_ws,
                              size_t ws_size, hipStream_t stream) {
  (void)in_sizes; (void)n_in; (void)out_size; (void)ws_size;
  const float* q    = (const float*)d_in[0];
  const float* k    = (const float*)d_in[1];
  const float* v    = (const float*)d_in[2];
  const float* mask = (const float*)d_in[3];

  float* out  = (float*)d_out;                              // [K,B,H,S,D]
  float* attn = out + (size_t)NMASK * NBH * SEQ * HD;       // [B,H,S,S]
  _Float16* vth = (_Float16*)d_ws;                          // 4 MB scratch

  vth_convert_kernel<<<NBH * 16, 256, 0, stream>>>(v, vth);
  attn_kernel<<<NBH * (SEQ / ROWS), 256, 0, stream>>>(q, k, mask, vth, out,
                                                      attn);
}
